// EMAVectorQuantizer_9225589752498
// MI455X (gfx1250) — matrix operand order repaired and checked by reading, not compiled, not run on hardware
//
#include <hip/hip_runtime.h>
#include <hip/hip_bf16.h>

// ---------------------------------------------------------------------------
// EMA Vector Quantizer forward for MI455X (gfx1250, wave32, WMMA)
//   z:      [64, 512, 256] f32   -> z_flat [N=32768, D=256]
//   weight: [K=1024, D=256] f32
// outputs (concatenated in d_out, f32):
//   [0]                      loss
//   [1 .. 1+N*D)             z_q_st  (== weight[argmin] numerically)
//   [1+N*D .. 1+N*D+N)       indices (as float)
//   [1+N*D+N]                perplexity
// ---------------------------------------------------------------------------

#define NTOK   1024      // codebook size K
#define CDIM   256       // embedding dim D
#define NROWS  32768     // N = 64*512
#define BETA   0.25f

typedef __attribute__((ext_vector_type(16))) __bf16 v16bf;
typedef __attribute__((ext_vector_type(8)))  float  v8f;

union BfFrag { v16bf v; unsigned short u[16]; };

__device__ __forceinline__ unsigned short f2bf_rne(float f) {
    unsigned u = __float_as_uint(f);
    unsigned r = u + 0x7FFFu + ((u >> 16) & 1u);   // round-to-nearest-even
    return (unsigned short)(r >> 16);
}
__device__ __forceinline__ float bf2f(unsigned short h) {
    return __uint_as_float(((unsigned)h) << 16);
}

// -------------------------------- zero accumulators ------------------------
__global__ void vq_zero_kernel(unsigned* counts, float* loss_sum) {
    int t = threadIdx.x;
    if (t < NTOK) counts[t] = 0u;
    if (t == 0)   loss_sum[0] = 0.0f;
}

// ---------------- split weight into bf16 hi/lo + row norms -----------------
// grid: NTOK blocks x CDIM threads
__global__ void vq_prep_kernel(const float* __restrict__ weight,
                               unsigned short* __restrict__ w_hi,
                               unsigned short* __restrict__ w_lo,
                               float* __restrict__ w_norm) {
    __shared__ float red[CDIM];
    const int r = blockIdx.x;
    const int t = threadIdx.x;
    const float w = weight[(size_t)r * CDIM + t];
    const unsigned short hi = f2bf_rne(w);
    const unsigned short lo = f2bf_rne(w - bf2f(hi));
    w_hi[(size_t)r * CDIM + t] = hi;
    w_lo[(size_t)r * CDIM + t] = lo;
    red[t] = w * w;
    __syncthreads();
    for (int s = CDIM / 2; s > 0; s >>= 1) {
        if (t < s) red[t] += red[t + s];
        __syncthreads();
    }
    if (t == 0) w_norm[r] = red[0];
}

// ------------------------------ main kernel --------------------------------
// 256 blocks x 256 threads (8 waves). Each wave owns a 16-row tile of z.
// Distance argmin uses  d = ||w||^2 - 2 * (z . w)   (||z||^2 is row-constant).
// z.w computed with split-bf16 WMMA: hi*hi + hi*lo + lo*hi (~fp32 accurate).
__global__ void __launch_bounds__(256)
vq_main_kernel(const float* __restrict__ z,
               const float* __restrict__ weight,
               const unsigned short* __restrict__ w_hi,
               const unsigned short* __restrict__ w_lo,
               const float* __restrict__ w_norm,
               unsigned* __restrict__ counts,
               float* __restrict__ loss_sum,
               float* __restrict__ out_zq,
               float* __restrict__ out_idx) {
    __shared__ int s_idx[128];

    const int lane  = threadIdx.x & 31;
    const int wid   = threadIdx.x >> 5;
    const int mbase = blockIdx.x * 128 + wid * 16;

    // ---- build A fragments (this wave's 16 z-rows, bf16 hi/lo) ----
    // A as loaded here (16x32): lane<16 holds K=0..15, lane>=16 holds
    // K=16..31 of row M = lane%16 (contiguous, matching B).
    const int arow = mbase + (lane & 15);
    const float* zrow = z + (size_t)arow * CDIM;
    const int a0 = (lane < 16) ? 0 : 16;

    BfFrag ah[8], al[8];
#pragma unroll
    for (int c = 0; c < 8; ++c) {
#pragma unroll
        for (int i = 0; i < 8; ++i) {
            float x0 = zrow[c * 32 + a0 + i];
            float x1 = zrow[c * 32 + a0 + 8 + i];
            unsigned short h0 = f2bf_rne(x0);
            unsigned short h1 = f2bf_rne(x1);
            ah[c].u[i]     = h0;  al[c].u[i]     = f2bf_rne(x0 - bf2f(h0));
            ah[c].u[8 + i] = h1;  al[c].u[8 + i] = f2bf_rne(x1 - bf2f(h1));
        }
    }

    // ---- sweep all 1024 codes in 64 chunks of 16 columns ----
    // B as loaded here (32x16 bf16): per-lane fragment = 16 contiguous K-values of
    // column n = lane%16  ->  32 contiguous bytes of w_hi/w_lo. No transpose.
    const int koff = (lane < 16) ? 0 : 16;
    float bestd[8];
    int   bidx[8];
#pragma unroll
    for (int j = 0; j < 8; ++j) { bestd[j] = __builtin_inff(); bidx[j] = 0; }

    for (int tc = 0; tc < 64; ++tc) {
        const int token = tc * 16 + (lane & 15);
        const unsigned short* wh = w_hi + (size_t)token * CDIM + koff;
        const unsigned short* wl = w_lo + (size_t)token * CDIM + koff;
        v8f acc = {};
#pragma unroll
        for (int c = 0; c < 8; ++c) {
            v16bf bh = *(const v16bf*)(wh + c * 32);
            v16bf bl = *(const v16bf*)(wl + c * 32);
            acc = __builtin_amdgcn_wmma_f32_16x16x32_bf16(
                      false, ah[c].v, false, bh, (short)0, acc, false, false);
            acc = __builtin_amdgcn_wmma_f32_16x16x32_bf16(
                      false, ah[c].v, false, bl, (short)0, acc, false, false);
            acc = __builtin_amdgcn_wmma_f32_16x16x32_bf16(
                      false, al[c].v, false, bh, (short)0, acc, false, false);
        }
        const float wn = w_norm[token];
#pragma unroll
        for (int j = 0; j < 8; ++j) {
            float dv = wn - 2.0f * acc[j];
            if (dv < bestd[j]) { bestd[j] = dv; bidx[j] = token; }
        }
    }

    // ---- cross-lane argmin within each 16-lane half ----
    // C layout: lanes 0-15 hold rows 0-7 (VGPR j = row j),
    //           lanes 16-31 hold rows 8-15.
#pragma unroll
    for (int m = 1; m < 16; m <<= 1) {
#pragma unroll
        for (int j = 0; j < 8; ++j) {
            float od = __shfl_xor(bestd[j], m, 32);
            int   oi = __shfl_xor(bidx[j],  m, 32);
            if (od < bestd[j] || (od == bestd[j] && oi < bidx[j])) {
                bestd[j] = od; bidx[j] = oi;
            }
        }
    }
    if ((lane & 15) == 0) {
        const int rbase = wid * 16 + ((lane < 16) ? 0 : 8);
#pragma unroll
        for (int j = 0; j < 8; ++j) s_idx[rbase + j] = bidx[j];
    }
    __syncthreads();

    // ---- gather z_q, commitment loss, histogram, index output ----
    float lsum = 0.0f;
    for (int j = 0; j < 16; ++j) {
        const int row = mbase + j;
        const int idx = s_idx[wid * 16 + j];
        const float* wrow = weight + (size_t)idx * CDIM;
        const float* zr   = z      + (size_t)row * CDIM;
        float*       orow = out_zq + (size_t)row * CDIM;
        for (int d = lane; d < CDIM; d += 32) {
            float wv = wrow[d];
            float zv = zr[d];
            orow[d] = wv;                 // z_q_st == z_q numerically
            float df = wv - zv;
            lsum += df * df;
        }
        if (lane == 0) {
            atomicAdd(&counts[idx], 1u);
            out_idx[row] = (float)idx;
        }
    }
#pragma unroll
    for (int m = 1; m < 32; m <<= 1) lsum += __shfl_xor(lsum, m, 32);
    if (lane == 0) atomicAdd(loss_sum, lsum);
}

// ------------------------- loss scale + perplexity -------------------------
__global__ void vq_finalize_kernel(const unsigned* __restrict__ counts,
                                   const float* __restrict__ loss_sum,
                                   float* __restrict__ out_loss,
                                   float* __restrict__ out_perp) {
    __shared__ float red[NTOK];
    const int t = threadIdx.x;
    float e = (float)counts[t] * (1.0f / (float)NROWS);
    red[t] = e * logf(e + 1e-10f);
    __syncthreads();
    for (int s = NTOK / 2; s > 0; s >>= 1) {
        if (t < s) red[t] += red[t + s];
        __syncthreads();
    }
    if (t == 0) {
        out_perp[0] = expf(-red[0]);
        out_loss[0] = BETA * loss_sum[0] * (1.0f / ((float)NROWS * (float)CDIM));
    }
}

// ---------------------------------------------------------------------------
extern "C" void kernel_launch(void* const* d_in, const int* in_sizes, int n_in,
                              void* d_out, int out_size, void* d_ws, size_t ws_size,
                              hipStream_t stream) {
    (void)in_sizes; (void)n_in; (void)out_size; (void)ws_size;

    const float* z      = (const float*)d_in[0];   // [32768, 256]
    const float* weight = (const float*)d_in[1];   // [1024, 256]

    // workspace layout
    unsigned short* w_hi  = (unsigned short*)d_ws;            // 512 KB
    unsigned short* w_lo  = w_hi + (size_t)NTOK * CDIM;       // 512 KB
    float*          w_nrm = (float*)(w_lo + (size_t)NTOK * CDIM);
    unsigned*       counts   = (unsigned*)(w_nrm + NTOK);
    float*          loss_sum = (float*)(counts + NTOK);

    // output layout (tuple concatenated flat)
    float* out      = (float*)d_out;
    float* out_loss = out;
    float* out_zq   = out + 1;
    float* out_idx  = out + 1 + (size_t)NROWS * CDIM;
    float* out_perp = out + 1 + (size_t)NROWS * CDIM + NROWS;

    vq_zero_kernel<<<1, NTOK, 0, stream>>>(counts, loss_sum);
    vq_prep_kernel<<<NTOK, CDIM, 0, stream>>>(weight, w_hi, w_lo, w_nrm);
    vq_main_kernel<<<NROWS / 128, 256, 0, stream>>>(
        z, weight, w_hi, w_lo, w_nrm, counts, loss_sum, out_zq, out_idx);
    vq_finalize_kernel<<<1, NTOK, 0, stream>>>(counts, loss_sum, out_loss, out_perp);
}
